// RegionProposalNetwork_12610023981674
// MI455X (gfx1250) — compile-verified, matrix operand order checked by reading, not run on hardware
//
#include <hip/hip_runtime.h>
#include <hip/hip_bf16.h>
#include <math.h>

typedef __attribute__((ext_vector_type(16))) _Float16 v16h;
typedef __attribute__((ext_vector_type(8)))  _Float16 v8h;
typedef __attribute__((ext_vector_type(2)))  _Float16 v2h;
typedef __attribute__((ext_vector_type(8)))  float    v8f;

#define NIMG   8
#define CCH    256
#define HF     64
#define WF     64
#define POS    4096       // HF*WF
#define AANCH  9
#define MANCH  36864      // POS*AANCH
#define GT     20
#define KDIM   2304       // CCH*9
#define KTILE  32
#define MTILE  128
#define NTILE  128
#define KPAD   40         // KTILE+8 halfs -> 80B row stride (16B aligned)
#define TOPN   1000
#define NCHUNK 144        // MANCH/256
#define IMGSZ  1024.0f
#define BBOXCLAMP 4.135166556742356f   // log(1000/16)
#define NMS_TH 0.7f
#define SBETA  (1.0f/9.0f)
#define NEGINF (-__builtin_inff())

// ---------------------------------------------------------------------------
// Load one 16x32 (MxK) 16-bit WMMA fragment for this lane from a K-major LDS
// row. Per ISA 7.12.2, lane (hsel,l16) holds two contiguous 8-half runs at
// K = hsel*8 and K = 16+hsel*8  ->  two ds_load_b128, no sub-word shuffling.
// ---------------------------------------------------------------------------
__device__ __forceinline__ v16h load_frag(const _Float16* row, int hsel)
{
    v8h lo = *(const v8h*)(row + hsel * 8);
    v8h hi = *(const v8h*)(row + 16 + hsel * 8);
    return __builtin_shufflevector(lo, hi, 0, 1, 2, 3, 4, 5, 6, 7,
                                   8, 9, 10, 11, 12, 13, 14, 15);
}

// ---------------------------------------------------------------------------
// Kernel 1: 3x3 conv (256->256) + bias + ReLU as implicit GEMM via WMMA f16.
// Per image: D[4096 x 256] = im2col(feat)[4096 x 2304] * W[2304 x 256].
// Block = 256 threads (8 wave32). Block tile 128(M) x 128(N); wave grid 4x2,
// each wave computes a 32x64 tile = 2x4 fragments -> 8 WMMAs per K-step.
// Both LDS tiles are K-major so fragments load as b128s.
// Output written NHWC: t[n][pos][co].
// ---------------------------------------------------------------------------
__global__ void __launch_bounds__(256)
conv3x3_relu_wmma(const float* __restrict__ feat, const float* __restrict__ cw,
                  const float* __restrict__ cb, float* __restrict__ t)
{
    __shared__ _Float16 As[MTILE][KPAD];   // [m][k]
    __shared__ _Float16 Bs[NTILE][KPAD];   // [co][k]

    const int n     = blockIdx.y;
    const int mt    = blockIdx.x & 31;          // 32 M-tiles
    const int nt    = blockIdx.x >> 5;          // 2  N-tiles
    const int tid   = threadIdx.x;
    const int lane  = tid & 31;
    const int wv    = tid >> 5;                 // 0..7
    const int wm    = wv & 3;                   // 4 waves along M (32 rows each)
    const int wn    = wv >> 2;                  // 2 waves along N (64 cols each)
    const int mBase = mt * MTILE;
    const int nBase = nt * NTILE;

    v8f zero8 = {0.f,0.f,0.f,0.f,0.f,0.f,0.f,0.f};
    v8f acc[2][4];
    #pragma unroll
    for (int i = 0; i < 2; ++i)
        #pragma unroll
        for (int j = 0; j < 4; ++j) acc[i][j] = zero8;

    for (int kt = 0; kt < KDIM; kt += KTILE) {
        // --- stage A tile (im2col, zero-padded borders), pair-packed ---
        for (int idx = tid; idx < MTILE * KTILE / 2; idx += 256) {
            int ml = idx >> 4;                 // 0..127
            int kk = (idx & 15) << 1;          // even K
            int m  = mBase + ml;
            int h  = m >> 6, x = m & 63;
            v2h pk;
            #pragma unroll
            for (int u = 0; u < 2; ++u) {
                int k  = kt + kk + u;
                int ci = k / 9, r = k - ci * 9;
                int kh = r / 3, kw = r - kh * 3;
                int ih = h + kh - 1, iw = x + kw - 1;
                float v = 0.0f;
                if ((unsigned)ih < 64u && (unsigned)iw < 64u)
                    v = feat[(((size_t)n * CCH + ci) * HF + ih) * WF + iw];
                pk[u] = (_Float16)v;
            }
            *(v2h*)&As[ml][kk] = pk;
        }
        // --- stage B tile: cw is [co][k] contiguous in k -> float2 loads ---
        for (int idx = tid; idx < NTILE * KTILE / 2; idx += 256) {
            int cl = idx >> 4;                 // 0..127
            int kk = (idx & 15) << 1;
            int co = nBase + cl;
            float2 w2 = *(const float2*)(cw + (size_t)co * KDIM + kt + kk);
            v2h pk; pk[0] = (_Float16)w2.x; pk[1] = (_Float16)w2.y;
            *(v2h*)&Bs[cl][kk] = pk;
        }
        __syncthreads();

        const int hsel = lane >> 4;
        const int l16  = lane & 15;
        v16h af[2], bf[4];
        #pragma unroll
        for (int f = 0; f < 2; ++f)
            af[f] = load_frag(&As[wm * 32 + f * 16 + l16][0], hsel);
        #pragma unroll
        for (int f = 0; f < 4; ++f)
            bf[f] = load_frag(&Bs[wn * 64 + f * 16 + l16][0], hsel);
        #pragma unroll
        for (int fm = 0; fm < 2; ++fm)
            #pragma unroll
            for (int fn = 0; fn < 4; ++fn)
                acc[fm][fn] = __builtin_amdgcn_wmma_f32_16x16x32_f16(
                    false, af[fm], false, bf[fn], (short)0, acc[fm][fn],
                    false, false);
        __syncthreads();
    }

    // --- epilogue: bias + ReLU, write NHWC ---
    const int hsel = lane >> 4;
    const int l16  = lane & 15;
    #pragma unroll
    for (int fm = 0; fm < 2; ++fm)
        #pragma unroll
        for (int fn = 0; fn < 4; ++fn)
            #pragma unroll
            for (int r = 0; r < 8; ++r) {
                int mrow = mBase + wm * 32 + fm * 16 + hsel * 8 + r;
                int col  = nBase + wn * 64 + fn * 16 + l16;
                float vo = acc[fm][fn][r] + cb[col];
                vo = fmaxf(vo, 0.0f);
                t[((size_t)n * POS + mrow) * CCH + col] = vo;
            }
}

// ---------------------------------------------------------------------------
// Kernel 2: 1x1 heads. One thread per (image,pos,output); 45 outputs/pos.
// t is NHWC so the 256-wide reduction is over contiguous memory.
// ---------------------------------------------------------------------------
__global__ void __launch_bounds__(256)
heads_1x1(const float* __restrict__ t, const float* __restrict__ clsw,
          const float* __restrict__ clsb, const float* __restrict__ regw,
          const float* __restrict__ regb, float* __restrict__ obj,
          float* __restrict__ deltas)
{
    int gid = blockIdx.x * 256 + threadIdx.x;
    if (gid >= NIMG * POS * 45) return;
    int o  = gid % 45;
    int pm = gid / 45;                       // n*POS + pos
    const float* tp = t + (size_t)pm * CCH;
    const float* wp; float b;
    if (o < 9) { wp = clsw + (size_t)o * CCH;        b = clsb[o];     }
    else       { wp = regw + (size_t)(o - 9) * CCH;  b = regb[o - 9]; }
    float s = b;
    for (int c = 0; c < CCH; ++c) s += tp[c] * wp[c];
    int n = pm / POS, pos = pm % POS;
    if (o < 9) {
        obj[(size_t)n * MANCH + pos * 9 + o] = s;
    } else {
        int a = (o - 9) >> 2, j = (o - 9) & 3;
        deltas[(((size_t)n * MANCH + (size_t)pos * 9 + a) << 2) + j] = s;
    }
}

// ---------------------------------------------------------------------------
// Kernel 3: decode + clip + small-box filter + sigmoid score.
// ---------------------------------------------------------------------------
__global__ void __launch_bounds__(256)
decode_clip(const float* __restrict__ obj, const float* __restrict__ deltas,
            const float* __restrict__ anchors, float* __restrict__ boxes,
            float* __restrict__ scores)
{
    int gid = blockIdx.x * 256 + threadIdx.x;
    if (gid >= NIMG * MANCH) return;
    const float* an = anchors + (size_t)gid * 4;
    float wa  = an[2] - an[0], ha = an[3] - an[1];
    float cxa = an[0] + 0.5f * wa, cya = an[1] + 0.5f * ha;
    const float* d = deltas + (size_t)gid * 4;
    float dw = fminf(d[2], BBOXCLAMP), dh = fminf(d[3], BBOXCLAMP);
    float cx = d[0] * wa + cxa, cy = d[1] * ha + cya;
    float w  = expf(dw) * wa,   h  = expf(dh) * ha;
    float x1 = fminf(fmaxf(cx - 0.5f * w, 0.f), IMGSZ);
    float y1 = fminf(fmaxf(cy - 0.5f * h, 0.f), IMGSZ);
    float x2 = fminf(fmaxf(cx + 0.5f * w, 0.f), IMGSZ);
    float y2 = fminf(fmaxf(cy + 0.5f * h, 0.f), IMGSZ);
    boxes[gid*4+0] = x1; boxes[gid*4+1] = y1;
    boxes[gid*4+2] = x2; boxes[gid*4+3] = y2;
    bool small = ((x2 - x1) < 0.01f) || ((y2 - y1) < 0.01f);
    float p = 1.0f / (1.0f + expf(-obj[gid]));
    scores[gid] = small ? NEGINF : p;
}

// ---------------------------------------------------------------------------
// Kernel 4: greedy NMS, one block per image, TOPN serial iterations of
// block-wide argmax (tie-break: smallest index, matching jnp.argmax).
// Writes the proposals section of d_out.
// ---------------------------------------------------------------------------
__global__ void __launch_bounds__(256)
nms_topk(const float* __restrict__ boxes_all, float* __restrict__ scores_all,
         float* __restrict__ out)
{
    int n = blockIdx.x, tid = threadIdx.x;
    const float* bx = boxes_all + (size_t)n * MANCH * 4;
    float* sc = scores_all + (size_t)n * MANCH;
    float* op = out + (size_t)n * TOPN * 4;
    __shared__ float smax[256]; __shared__ int simax[256];
    __shared__ float sel[5];    // x1,y1,x2,y2,area

    for (int it = 0; it < TOPN; ++it) {
        float lm = NEGINF; int li = 0;
        for (int m = tid; m < MANCH; m += 256) {
            float v = sc[m];
            if (v > lm) { lm = v; li = m; }
        }
        smax[tid] = lm; simax[tid] = li;
        __syncthreads();
        for (int s = 128; s > 0; s >>= 1) {
            if (tid < s) {
                float a = smax[tid], b = smax[tid + s];
                int ia = simax[tid], ib = simax[tid + s];
                if (b > a || (b == a && ib < ia)) { smax[tid] = b; simax[tid] = ib; }
            }
            __syncthreads();
        }
        if (tid == 0) {
            int j = simax[0]; float s0 = smax[0];
            float x1 = bx[j*4], y1 = bx[j*4+1], x2 = bx[j*4+2], y2 = bx[j*4+3];
            bool valid = (s0 > -1.0e30f);
            op[it*4+0] = valid ? x1 : 0.f; op[it*4+1] = valid ? y1 : 0.f;
            op[it*4+2] = valid ? x2 : 0.f; op[it*4+3] = valid ? y2 : 0.f;
            sel[0] = x1; sel[1] = y1; sel[2] = x2; sel[3] = y2;
            sel[4] = (x2 - x1) * (y2 - y1);
            sc[j] = NEGINF;
        }
        __syncthreads();
        float jx1 = sel[0], jy1 = sel[1], jx2 = sel[2], jy2 = sel[3], ja = sel[4];
        for (int m = tid; m < MANCH; m += 256) {
            float v = sc[m];
            if (v <= -1.0e30f) continue;
            float mx1 = bx[m*4], my1 = bx[m*4+1], mx2 = bx[m*4+2], my2 = bx[m*4+3];
            float iw = fmaxf(0.f, fminf(jx2, mx2) - fmaxf(jx1, mx1));
            float ih = fmaxf(0.f, fminf(jy2, my2) - fmaxf(jy1, my1));
            float inter = iw * ih;
            float iou = inter / (ja + (mx2 - mx1) * (my2 - my1) - inter);
            if (iou > NMS_TH) sc[m] = NEGINF;
        }
        __syncthreads();
    }
}

// ---------------------------------------------------------------------------
// Matcher helpers.
// ---------------------------------------------------------------------------
__device__ __forceinline__ float box_iou(const float* g, float ax1, float ay1,
                                         float ax2, float ay2, float area_a)
{
    float gx1 = g[0], gy1 = g[1], gx2 = g[2], gy2 = g[3];
    float area_g = (gx2 - gx1) * (gy2 - gy1);
    float iw = fmaxf(0.f, fminf(gx2, ax2) - fmaxf(gx1, ax1));
    float ih = fmaxf(0.f, fminf(gy2, ay2) - fmaxf(gy1, ay1));
    float inter = iw * ih;
    return inter / (area_g + area_a - inter);
}

__global__ void init_k(unsigned* gbits, float* sums, int* selc)
{
    int i = threadIdx.x;
    if (i < NIMG * GT) gbits[i] = 0u;
    if (i < 2) { sums[i] = 0.f; selc[i] = 0; }
}

// Kernel 5a: best IoU / argmax over gts per anchor + per-gt max over anchors.
__global__ void __launch_bounds__(256)
match_pass1(const float* __restrict__ gtb, const float* __restrict__ anchors,
            float* __restrict__ biou, int* __restrict__ argm,
            unsigned* __restrict__ gbits)
{
    int gid = blockIdx.x * 256 + threadIdx.x;
    if (gid >= NIMG * MANCH) return;
    int n = gid / MANCH;
    const float* an = anchors + (size_t)gid * 4;
    float ax1 = an[0], ay1 = an[1], ax2 = an[2], ay2 = an[3];
    float aa = (ax2 - ax1) * (ay2 - ay1);
    float best = -1.0f; int bi = 0;
    for (int g = 0; g < GT; ++g) {
        float iou = box_iou(gtb + ((size_t)n * GT + g) * 4, ax1, ay1, ax2, ay2, aa);
        if (iou > best) { best = iou; bi = g; }
        atomicMax(&gbits[n * GT + g], __float_as_uint(iou)); // iou>=0: uint order ok
    }
    biou[gid] = best; argm[gid] = bi;
}

// Kernel 5b: thresholds + low-quality matches -> labels + encoded targets.
__global__ void __launch_bounds__(256)
match_pass2(const float* __restrict__ gtb, const float* __restrict__ anchors,
            const float* __restrict__ biou, const int* __restrict__ argm,
            const unsigned* __restrict__ gbits, float* __restrict__ labels,
            float* __restrict__ targets)
{
    int gid = blockIdx.x * 256 + threadIdx.x;
    if (gid >= NIMG * MANCH) return;
    int n = gid / MANCH;
    const float* an = anchors + (size_t)gid * 4;
    float ax1 = an[0], ay1 = an[1], ax2 = an[2], ay2 = an[3];
    float aa = (ax2 - ax1) * (ay2 - ay1);
    bool lq = false;
    for (int g = 0; g < GT; ++g) {
        float iou = box_iou(gtb + ((size_t)n * GT + g) * 4, ax1, ay1, ax2, ay2, aa);
        if (__float_as_uint(iou) == gbits[n * GT + g]) lq = true;
    }
    float bi = biou[gid];
    int am = argm[gid];
    int mm = (bi >= 0.7f) ? am : -2;
    if (bi < 0.3f) mm = -1;            // NEG_LOW == 0.0, iou >= 0 always
    if (lq) mm = am;
    labels[gid] = (mm >= 0) ? 1.f : ((mm == -1) ? 0.f : -1.f);
    int gs = (mm >= 0) ? mm : 0;       // clip(matched, 0)
    const float* g = gtb + ((size_t)n * GT + gs) * 4;
    float wa = ax2 - ax1, ha = ay2 - ay1;
    float cxa = ax1 + 0.5f * wa, cya = ay1 + 0.5f * ha;
    float wg = g[2] - g[0], hg = g[3] - g[1];
    float cxg = g[0] + 0.5f * wg, cyg = g[1] + 0.5f * hg;
    targets[gid*4+0] = (cxg - cxa) / wa;
    targets[gid*4+1] = (cyg - cya) / ha;
    targets[gid*4+2] = logf(wg / wa);
    targets[gid*4+3] = logf(hg / ha);
}

// ---------------------------------------------------------------------------
// Balanced sampler (deterministic index-ranked stand-in for randperm) + losses.
// ---------------------------------------------------------------------------
__global__ void __launch_bounds__(256)
samp_count(const float* __restrict__ labels, int* __restrict__ cntp,
           int* __restrict__ cntn)
{
    int n = blockIdx.x / NCHUNK, chunk = blockIdx.x % NCHUNK;
    size_t idx = (size_t)n * MANCH + chunk * 256 + threadIdx.x;
    float lab = labels[idx];
    int cp = __syncthreads_count(lab == 1.0f);
    int cn = __syncthreads_count(lab == 0.0f);
    if (threadIdx.x == 0) { cntp[n*NCHUNK+chunk] = cp; cntn[n*NCHUNK+chunk] = cn; }
}

__global__ void samp_scan(const int* __restrict__ cntp, const int* __restrict__ cntn,
                          int* __restrict__ basep, int* __restrict__ basen,
                          int* __restrict__ nump, int* __restrict__ numn)
{
    int n = threadIdx.x;
    if (n >= NIMG) return;
    int ap = 0, an = 0;
    for (int c = 0; c < NCHUNK; ++c) {
        basep[n*NCHUNK+c] = ap; ap += cntp[n*NCHUNK+c];
        basen[n*NCHUNK+c] = an; an += cntn[n*NCHUNK+c];
    }
    int np = (ap < 128) ? ap : 128;
    int nn = (an < 256 - np) ? an : (256 - np);
    nump[n] = np; numn[n] = nn;
}

__global__ void __launch_bounds__(256)
samp_loss(const float* __restrict__ labels, const float* __restrict__ obj,
          const float* __restrict__ deltas, const float* __restrict__ targets,
          const int* __restrict__ basep, const int* __restrict__ basen,
          const int* __restrict__ nump, const int* __restrict__ numn,
          float* __restrict__ sums, int* __restrict__ selc)
{
    int n = blockIdx.x / NCHUNK, chunk = blockIdx.x % NCHUNK;
    int tid = threadIdx.x;
    size_t idx = (size_t)n * MANCH + chunk * 256 + tid;
    float lab = labels[idx];
    bool pp = (lab == 1.0f), pn = (lab == 0.0f);
    int lane = tid & 31, wv = tid >> 5;
    unsigned bp = (unsigned)__ballot(pp);
    unsigned bn = (unsigned)__ballot(pn);
    unsigned mlow = (1u << lane) - 1u;
    int lp = __popc(bp & mlow), ln = __popc(bn & mlow);
    __shared__ int wtp[8], wtn[8];
    if (lane == 0) { wtp[wv] = __popc(bp); wtn[wv] = __popc(bn); }
    __syncthreads();
    int wbp = 0, wbn = 0;
    for (int i = 0; i < wv; ++i) { wbp += wtp[i]; wbn += wtn[i]; }
    int rp = basep[n*NCHUNK+chunk] + wbp + lp;
    int rn = basen[n*NCHUNK+chunk] + wbn + ln;
    bool sel_pos = pp && (rp < nump[n]);
    bool sel_neg = pn && (rn < numn[n]);
    if (sel_pos || sel_neg) {
        float o = obj[idx];
        float y = sel_pos ? 1.0f : 0.0f;
        float bce = fmaxf(o, 0.f) - o * y + log1pf(expf(-fabsf(o)));
        atomicAdd(&sums[0], bce);
        atomicAdd(&selc[0], 1);
    }
    if (sel_pos) {
        float s = 0.f;
        for (int j = 0; j < 4; ++j) {
            float d = fabsf(deltas[idx*4+j] - targets[idx*4+j]);
            s += (d < SBETA) ? (0.5f * d * d / SBETA) : (d - 0.5f * SBETA);
        }
        atomicAdd(&sums[1], s);
        atomicAdd(&selc[1], 1);
    }
}

__global__ void finalize_k(const float* __restrict__ sums,
                           const int* __restrict__ selc, float* __restrict__ out)
{
    int dn = (selc[0] > 1) ? selc[0] : 1;
    int dl = (4 * selc[1] > 1) ? 4 * selc[1] : 1;
    out[NIMG * TOPN * 4 + 0] = sums[0] / (float)dn;
    out[NIMG * TOPN * 4 + 1] = sums[1] / (float)dl;
}

// ---------------------------------------------------------------------------
extern "C" void kernel_launch(void* const* d_in, const int* in_sizes, int n_in,
                              void* d_out, int out_size, void* d_ws, size_t ws_size,
                              hipStream_t stream)
{
    (void)in_sizes; (void)n_in; (void)out_size; (void)ws_size;
    const float* feat    = (const float*)d_in[0];
    const float* convw   = (const float*)d_in[1];
    const float* convb   = (const float*)d_in[2];
    const float* clsw    = (const float*)d_in[3];
    const float* clsb    = (const float*)d_in[4];
    const float* regw    = (const float*)d_in[5];
    const float* regb    = (const float*)d_in[6];
    const float* anchors = (const float*)d_in[7];
    const float* gtb     = (const float*)d_in[8];
    float* out = (float*)d_out;

    // workspace layout (floats then ints)
    float* fb      = (float*)d_ws;
    float* t       = fb;                                   // NHWC conv output
    float* obj     = t       + (size_t)NIMG * POS * CCH;
    float* deltas  = obj     + (size_t)NIMG * MANCH;
    float* boxes   = deltas  + (size_t)NIMG * MANCH * 4;
    float* scores  = boxes   + (size_t)NIMG * MANCH * 4;
    float* biou    = scores  + (size_t)NIMG * MANCH;
    float* labels  = biou    + (size_t)NIMG * MANCH;
    float* targets = labels  + (size_t)NIMG * MANCH;
    float* sums    = targets + (size_t)NIMG * MANCH * 4;   // [0]=cls [1]=loc
    int*   argm    = (int*)(sums + 2);
    int*   cntp    = argm  + (size_t)NIMG * MANCH;
    int*   cntn    = cntp  + NIMG * NCHUNK;
    int*   basep   = cntn  + NIMG * NCHUNK;
    int*   basen   = basep + NIMG * NCHUNK;
    int*   nump    = basen + NIMG * NCHUNK;
    int*   numn    = nump  + NIMG;
    int*   selc    = numn  + NIMG;                          // [0]=sel [1]=pos
    unsigned* gbits = (unsigned*)(selc + 2);                // NIMG*GT

    init_k<<<1, 256, 0, stream>>>(gbits, sums, selc);

    dim3 gconv((POS / MTILE) * (CCH / NTILE), NIMG);        // (64, 8)
    conv3x3_relu_wmma<<<gconv, 256, 0, stream>>>(feat, convw, convb, t);

    int hthreads = NIMG * POS * 45;
    heads_1x1<<<(hthreads + 255) / 256, 256, 0, stream>>>(t, clsw, clsb, regw,
                                                          regb, obj, deltas);

    int am = NIMG * MANCH;
    decode_clip<<<(am + 255) / 256, 256, 0, stream>>>(obj, deltas, anchors,
                                                      boxes, scores);

    nms_topk<<<NIMG, 256, 0, stream>>>(boxes, scores, out);

    match_pass1<<<(am + 255) / 256, 256, 0, stream>>>(gtb, anchors, biou, argm,
                                                      gbits);
    match_pass2<<<(am + 255) / 256, 256, 0, stream>>>(gtb, anchors, biou, argm,
                                                      gbits, labels, targets);

    samp_count<<<NIMG * NCHUNK, 256, 0, stream>>>(labels, cntp, cntn);
    samp_scan<<<1, NIMG, 0, stream>>>(cntp, cntn, basep, basen, nump, numn);
    samp_loss<<<NIMG * NCHUNK, 256, 0, stream>>>(labels, obj, deltas, targets,
                                                 basep, basen, nump, numn,
                                                 sums, selc);
    finalize_k<<<1, 1, 0, stream>>>(sums, selc, out);
}
